// GraphLayer_40733469835302
// MI455X (gfx1250) — matrix operand order repaired and checked by reading, not compiled, not run on hardware
//
#include <hip/hip_runtime.h>
#include <hip/hip_bf16.h>

// ---------------------------------------------------------------------------
// MI455X (gfx1250) SuperGlue-style pipeline.
// All GEMM-shaped work (projections, attention, MLPs, 3x3 convs as implicit
// GEMM) runs through wave32 v_wmma_f32_16x16x32_bf16, 4 WMMAs per K-step.
// Staging uses CDNA5 async global->LDS copies (ASYNCcnt) where the copy is
// layout-preserving, and 16B vector loads + LDS transpose otherwise.
// ---------------------------------------------------------------------------

typedef __attribute__((ext_vector_type(16))) __bf16 v16bf;
typedef __attribute__((ext_vector_type(8)))  __bf16 bf8v;
typedef __attribute__((ext_vector_type(8)))  float  v8f;

#define NB   8
#define NPT  1024
#define HW2  65536   // 256*256
#define IMGW 256

// Low 32 bits of a generic LDS pointer == wave-relative LDS byte offset
// (ISA 10.2: flat->LDS address is addr[31:0]).
__device__ __forceinline__ unsigned lds_off(const void* p) {
  return (unsigned)(unsigned long long)p;
}
__device__ __forceinline__ void async_g2l_b128(unsigned loff, const void* g) {
  asm volatile("global_load_async_to_lds_b128 %0, %1, off"
               :: "v"(loff), "v"(g) : "memory");
}
__device__ __forceinline__ void wait_async() {
  asm volatile("s_wait_asynccnt 0x0" ::: "memory");
}

// ---------------------------------------------------------------------------
// Generic batched WMMA GEMM:  C = alpha*op(A)*op(B) (+bias)(*g+be)(relu)(+resid)
//  TA: A stored [K,M] (contiguous M)   else [M,K] (contiguous K)
//  TB: B stored [N,K] (contiguous K)   else [K,N] (contiguous N)
// Block = 128 threads / 4 waves; tile = 64(M) x 64(N) x 32(K); 4 acc/wave.
// Requires: M mult 64, N mult 64, K mult 32, row strides mult of 8 elements,
// bases 16B aligned (all call sites satisfy this; C store keeps guards).
// Batch z split zb=z/HZ, zh=z%HZ. C element (m,n) at m*ldc_m + n*ldc_n.
// ---------------------------------------------------------------------------
struct GemmP {
  const __bf16* A; const __bf16* Bm;
  float* Cf; __bf16* Cb;
  const float* bias; const float* gamma; const float* beta; const float* resid;
  int M, N, K, lda, ldb, ldc_m, ldc_n, HZ, relu;
  long long sAb, sAh, sBb, sBh, sCfb, sCbb, sCh, sRb;
  float alpha;
};

template <bool TA, bool TB>
__global__ void __launch_bounds__(128) k_gemm(GemmP p) {
  __shared__ __bf16 As[64][40];  // [m][k], rows padded to 80B
  __shared__ __bf16 Bs[64][40];  // [n][k]

  const int z  = blockIdx.z;
  const int zb = z / p.HZ, zh = z % p.HZ;
  const long long offA  = (long long)zb * p.sAb  + (long long)zh * p.sAh;
  const long long offB  = (long long)zb * p.sBb  + (long long)zh * p.sBh;
  const long long offCf = (long long)zb * p.sCfb + (long long)zh * p.sCh;
  const long long offCb = (long long)zb * p.sCbb + (long long)zh * p.sCh;
  const long long offR  = (long long)zb * p.sRb;

  const int m0 = blockIdx.y << 6;
  const int n0 = blockIdx.x << 6;
  const int t = threadIdx.x;
  const int w = t >> 5, lane = t & 31;
  const int lh = lane & 15, kh = lane >> 4;

  v8f acc[4];
#pragma unroll
  for (int h = 0; h < 4; ++h)
#pragma unroll
    for (int r = 0; r < 8; ++r) acc[h][r] = 0.f;

  for (int kc = 0; kc < p.K; kc += 32) {
    // ---- stage A tile 64x32 ----
    if (!TA) {
      // layout-preserving: async DMA, 16B per lane, 2 issues/thread
#pragma unroll
      for (int s = 0; s < 2; ++s) {
        int i = t + (s << 7);            // 0..255 chunks of 8 bf16
        int r = i >> 2, seg = i & 3;
        const __bf16* g = p.A + offA + (long long)(m0 + r) * p.lda + kc + (seg << 3);
        async_g2l_b128(lds_off(&As[r][seg << 3]), g);
      }
    } else {
      // A stored [K,M]: vector-load 8 m's, transpose into LDS
#pragma unroll
      for (int s = 0; s < 2; ++s) {
        int i = t + (s << 7);            // k(32) x mseg(8)
        int k = i >> 3, ms = i & 7;
        bf8v v = *(const bf8v*)(p.A + offA + (long long)(kc + k) * p.lda + m0 + (ms << 3));
#pragma unroll
        for (int e = 0; e < 8; ++e) As[(ms << 3) + e][k] = v[e];
      }
    }
    // ---- stage B tile 64(n) x 32(k) as [n][k] ----
    if (!TB) {
      if (kc + 32 < p.K)  // speculative prefetch of next B k-chunk
        __builtin_prefetch(p.Bm + offB + (long long)(kc + 32 + (t >> 2)) * p.ldb + n0, 0, 1);
#pragma unroll
      for (int s = 0; s < 2; ++s) {
        int i = t + (s << 7);            // k(32) x nseg(8)
        int k = i >> 3, ns = i & 7;
        bf8v v = *(const bf8v*)(p.Bm + offB + (long long)(kc + k) * p.ldb + n0 + (ns << 3));
#pragma unroll
        for (int e = 0; e < 8; ++e) Bs[(ns << 3) + e][k] = v[e];
      }
    } else {
      // B stored [N,K]: layout-preserving async DMA
#pragma unroll
      for (int s = 0; s < 2; ++s) {
        int i = t + (s << 7);            // n(64) x kseg(4)
        int n = i >> 2, ks = i & 3;
        const __bf16* g = p.Bm + offB + (long long)(n0 + n) * p.ldb + kc + (ks << 3);
        async_g2l_b128(lds_off(&Bs[n][ks << 3]), g);
      }
    }
    if (!TA || TB) wait_async();
    __syncthreads();

    // ---- fragments per ISA 16-bit A/B VGPR layouts (wave32) ----
    v16bf av;
    const int ml = (w << 4) + lh;
#pragma unroll
    for (int j = 0; j < 8; ++j) {
      av[j]     = As[ml][(kh << 3) + j];        // K = j + 8*kh
      av[j + 8] = As[ml][16 + (kh << 3) + j];   // K = 16 + j + 8*kh
    }
#pragma unroll
    for (int h = 0; h < 4; ++h) {
      v16bf bv;
#pragma unroll
      for (int j = 0; j < 16; ++j)
        bv[j] = Bs[(h << 4) + lh][(j & 8) * 2 + (kh << 3) + (j & 7)];  // K = 8*kh + j (j<8), 16 + 8*kh + j-8 (j>=8)
      acc[h] = __builtin_amdgcn_wmma_f32_16x16x32_bf16(
          false, av, false, bv, (short)0, acc[h], false, false);
    }
    __syncthreads();
  }

  // ---- epilogue: vgpr r -> M = m0+16w + r + 8*kh, N = n0 + 16*h + lh ----
#pragma unroll
  for (int h = 0; h < 4; ++h) {
    int n_g = n0 + (h << 4) + lh;
#pragma unroll
    for (int r = 0; r < 8; ++r) {
      int m_g = m0 + (w << 4) + r + (kh << 3);
      if (m_g < p.M && n_g < p.N) {
        float v = p.alpha * acc[h][r];
        if (p.bias)  v += p.bias[m_g];
        if (p.gamma) v = v * p.gamma[m_g] + p.beta[m_g];
        if (p.relu)  v = fmaxf(v, 0.f);
        long long ci = (long long)m_g * p.ldc_m + (long long)n_g * p.ldc_n;
        if (p.resid) v += p.resid[offR + ci];
        if (p.Cf) p.Cf[offCf + ci] = v;
        if (p.Cb) p.Cb[offCb + ci] = (__bf16)v;
      }
    }
  }
}

// ---------------------------------------------------------------------------
// Implicit-GEMM 3x3 conv (SAME). Weights repacked tap-major [9][CO][CIN] bf16.
// Block (128 thr / 4 waves) computes 64 out-channels x 64 pixels of one row.
// Weight tiles staged via async DMA. Image tiles: one thread per (pixel,
// 8-channel seg) -> single boundary guard, 8 strided loads, one b128 LDS store.
// ---------------------------------------------------------------------------
struct ConvP {
  const __bf16* in; const __bf16* wt;
  const float* bias; const float* gamma; const float* beta;
  float* outf; __bf16* outb;
  int CIN, CO, relu;
};

__global__ void __launch_bounds__(128) k_conv3(ConvP p) {
  __shared__ __bf16 As[64][40];
  __shared__ __bf16 Bs[64][40];

  const int x0 = blockIdx.x << 6;   // 64 pixels
  const int y  = blockIdx.y;
  const int copb = p.CO >> 6;
  const int b = blockIdx.z / copb;
  const int co0 = (blockIdx.z % copb) << 6;
  const int t = threadIdx.x;
  const int w = t >> 5, lane = t & 31;
  const int lh = lane & 15, kh = lane >> 4;

  v8f acc[4];
#pragma unroll
  for (int h = 0; h < 4; ++h)
#pragma unroll
    for (int r = 0; r < 8; ++r) acc[h][r] = 0.f;

  for (int tap = 0; tap < 9; ++tap) {
    const int dy = tap / 3 - 1, dx = tap % 3 - 1;
    const int yy = y + dy;
    for (int kc = 0; kc < p.CIN; kc += 32) {
      // weights: layout-preserving async DMA (64x32, 16B/lane, 2/thread)
#pragma unroll
      for (int s = 0; s < 2; ++s) {
        int i = t + (s << 7);
        int r = i >> 2, seg = i & 3;
        const __bf16* g = p.wt + ((long long)tap * p.CO + co0 + r) * p.CIN + kc + (seg << 3);
        async_g2l_b128(lds_off(&As[r][seg << 3]), g);
      }
      // image tile 64(px) x 32(c): thread owns (pixel, 8-channel seg);
      // single guard, channel-strided loads, aligned 16B LDS store
#pragma unroll
      for (int s = 0; s < 2; ++s) {
        int i = t + (s << 7);            // n(64) x kseg(4)
        int n = i >> 2, ks = i & 3;
        int xx = x0 + n + dx;
        bf8v v;
#pragma unroll
        for (int e = 0; e < 8; ++e) v[e] = (__bf16)0.0f;
        if (yy >= 0 && yy < IMGW && xx >= 0 && xx < IMGW) {
          const __bf16* base =
              p.in + ((long long)b * p.CIN + kc + (ks << 3)) * HW2 + yy * IMGW + xx;
#pragma unroll
          for (int e = 0; e < 8; ++e) v[e] = base[(long long)e * HW2];
        }
        *(bf8v*)(&Bs[n][ks << 3]) = v;
      }
      wait_async();
      __syncthreads();

      v16bf av;
      const int ml = (w << 4) + lh;
#pragma unroll
      for (int j = 0; j < 8; ++j) {
        av[j]     = As[ml][(kh << 3) + j];
        av[j + 8] = As[ml][16 + (kh << 3) + j];
      }
#pragma unroll
      for (int h = 0; h < 4; ++h) {
        v16bf bv;
#pragma unroll
        for (int j = 0; j < 16; ++j)
          bv[j] = Bs[(h << 4) + lh][(j & 8) * 2 + (kh << 3) + (j & 7)];
        acc[h] = __builtin_amdgcn_wmma_f32_16x16x32_bf16(
            false, av, false, bv, (short)0, acc[h], false, false);
      }
      __syncthreads();
    }
  }

#pragma unroll
  for (int h = 0; h < 4; ++h) {
    const int px = y * IMGW + x0 + (h << 4) + lh;
#pragma unroll
    for (int r = 0; r < 8; ++r) {
      int co = co0 + (w << 4) + r + (kh << 3);
      float v = acc[h][r] + p.bias[co];
      if (p.gamma) v = v * p.gamma[co] + p.beta[co];
      if (p.relu)  v = fmaxf(v, 0.f);
      long long o = ((long long)b * p.CO + co) * HW2 + px;
      if (p.outf) p.outf[o] = v;
      if (p.outb) p.outb[o] = (__bf16)v;
    }
  }
}

// ------------------------------ small kernels ------------------------------
__global__ void k_cvt(const float* s, __bf16* d, long long n) {
  long long i = (long long)blockIdx.x * blockDim.x + threadIdx.x;
  if (i < n) d[i] = (__bf16)s[i];
}

// dc_w [O][C][3][3] f32 -> [tap][O][C] bf16
__global__ void k_rearr(const float* s, __bf16* d, int O, int C) {
  long long n = (long long)9 * O * C;
  long long i = (long long)blockIdx.x * blockDim.x + threadIdx.x;
  if (i >= n) return;
  int tap = (int)(i / ((long long)O * C));
  int o = (int)((i / C) % O);
  int c = (int)(i % C);
  d[i] = (__bf16)s[((long long)o * C + c) * 9 + tap];
}

__global__ void k_zero32(unsigned* p, long long n) {
  long long i = (long long)blockIdx.x * blockDim.x + threadIdx.x;
  if (i < n) p[i] = 0u;
}

__global__ void k_gather(const float* fm, const int* kp, __bf16* g) {
  int i = blockIdx.x * blockDim.x + threadIdx.x;
  if (i >= NB * 128 * NPT) return;
  int n = i & (NPT - 1), c = (i >> 10) & 127, b = i >> 17;
  int j = (b << 10) + n;
  int idx = kp[j * 2] * IMGW + kp[j * 2 + 1];
  g[i] = (__bf16)fm[((long long)(b * 128 + c)) * HW2 + idx];
}

__global__ void k_keinput(const int* kp, __bf16* x0) {
  int i = blockIdx.x * blockDim.x + threadIdx.x;
  if (i >= NB * NPT) return;
  int n = i & (NPT - 1), b = i >> 10;
  const float inv = 1.0f / 179.2f;  // 0.7 * 256
  float k0 = (float)kp[i * 2], k1 = (float)kp[i * 2 + 1];
  x0[(b * 3 + 0) * NPT + n] = (__bf16)((k0 - 128.f) * inv);
  x0[(b * 3 + 1) * NPT + n] = (__bf16)((k1 - 128.f) * inv);
  x0[(b * 3 + 2) * NPT + n] = (__bf16)1.0f;
}

// tiny fused 1x1-conv + BN + relu (keypoint encoder layers, K<=64)
__global__ void k_lin_small(const __bf16* x, const float* w, const float* b,
                            const float* g, const float* be, __bf16* y,
                            int K, int O) {
  int i = blockIdx.x * blockDim.x + threadIdx.x;
  if (i >= NB * O * NPT) return;
  int n = i & (NPT - 1);
  int o = (i >> 10) % O;
  int bb = i / (O * NPT);
  float a = 0.f;
  for (int k = 0; k < K; ++k)
    a += w[o * K + k] * (float)x[(bb * K + k) * NPT + n];
  a += b[o];
  a = a * g[o] + be[o];
  y[(bb * O + o) * NPT + n] = (__bf16)fmaxf(a, 0.f);
}

// in-place row softmax over 1024 bf16 elements, one wave per row
__global__ void k_softmax(__bf16* pbuf) {
  int w = threadIdx.x >> 5, lane = threadIdx.x & 31;
  long long row = (long long)blockIdx.x * 8 + w;
  __bf16* r = pbuf + row * 1024;
  float v[32];
  float mx = -3.0e38f;
#pragma unroll
  for (int j = 0; j < 32; ++j) { v[j] = (float)r[lane + (j << 5)]; mx = fmaxf(mx, v[j]); }
  for (int o = 16; o; o >>= 1) mx = fmaxf(mx, __shfl_xor(mx, o, 32));
  float s = 0.f;
#pragma unroll
  for (int j = 0; j < 32; ++j) { v[j] = __expf(v[j] - mx); s += v[j]; }
  for (int o = 16; o; o >>= 1) s += __shfl_xor(s, o, 32);
  float inv = 1.f / s;
#pragma unroll
  for (int j = 0; j < 32; ++j) r[lane + (j << 5)] = (__bf16)(v[j] * inv);
}

// deterministic scatter: duplicate keypoint indices -> highest n wins
__global__ void k_scatter(const int* kp, const __bf16* d0, __bf16* zb) {
  int i = blockIdx.x * blockDim.x + threadIdx.x;
  if (i >= NB * NPT) return;
  int n = i & (NPT - 1), b = i >> 10;
  int idx = kp[i * 2] * IMGW + kp[i * 2 + 1];
  for (int n2 = n + 1; n2 < NPT; ++n2) {
    int j = (b << 10) + n2;
    if (kp[j * 2] * IMGW + kp[j * 2 + 1] == idx) return;  // later dup wins
  }
  for (int c = 0; c < 256; ++c)
    zb[((long long)(b * 256 + c) << 16) + idx] = d0[((b << 8) + c) * NPT + n];
}

// ------------------------------ host driver -------------------------------
static inline void gemm(hipStream_t s, const GemmP& p, int zc, int var) {
  dim3 g((p.N + 63) / 64, (p.M + 63) / 64, zc);
  if (var == 0)      k_gemm<false, false><<<g, 128, 0, s>>>(p);
  else if (var == 1) k_gemm<true,  false><<<g, 128, 0, s>>>(p);
  else               k_gemm<false, true ><<<g, 128, 0, s>>>(p);
}

extern "C" void kernel_launch(void* const* d_in, const int* in_sizes, int n_in,
                              void* d_out, int out_size, void* d_ws, size_t ws_size,
                              hipStream_t stream) {
  const float* feature = (const float*)d_in[0];
  const int*   kp      = (const int*)d_in[1];
  const float* fm_w = (const float*)d_in[2];
  const float* fm_b = (const float*)d_in[3];
  const float* ke_w0 = (const float*)d_in[4];  const float* ke_b0 = (const float*)d_in[5];
  const float* ke_w1 = (const float*)d_in[6];  const float* ke_b1 = (const float*)d_in[7];
  const float* ke_w2 = (const float*)d_in[8];  const float* ke_b2 = (const float*)d_in[9];
  const float* ke_w3 = (const float*)d_in[10]; const float* ke_b3 = (const float*)d_in[11];
  const float* ke_g0 = (const float*)d_in[12]; const float* ke_be0 = (const float*)d_in[13];
  const float* ke_g1 = (const float*)d_in[14]; const float* ke_be1 = (const float*)d_in[15];
  const float* ke_g2 = (const float*)d_in[16]; const float* ke_be2 = (const float*)d_in[17];
  const float* at_qw = (const float*)d_in[18]; const float* at_qb = (const float*)d_in[19];
  const float* at_kw = (const float*)d_in[20]; const float* at_kb = (const float*)d_in[21];
  const float* at_vw = (const float*)d_in[22]; const float* at_vb = (const float*)d_in[23];
  const float* at_mw = (const float*)d_in[24]; const float* at_mb = (const float*)d_in[25];
  const float* ml_w1 = (const float*)d_in[26]; const float* ml_b1 = (const float*)d_in[27];
  const float* ml_g1 = (const float*)d_in[28]; const float* ml_be1 = (const float*)d_in[29];
  const float* ml_w2 = (const float*)d_in[30]; const float* ml_b2 = (const float*)d_in[31];
  const float* fp_w = (const float*)d_in[32];  const float* fp_b = (const float*)d_in[33];
  const float* dc_w1 = (const float*)d_in[34]; const float* dc_b1 = (const float*)d_in[35];
  const float* dc_g  = (const float*)d_in[36]; const float* dc_be = (const float*)d_in[37];
  const float* dc_w2 = (const float*)d_in[38]; const float* dc_b2 = (const float*)d_in[39];

  char* ws = (char*)d_ws;
  size_t off = 0;
  auto alloc = [&](size_t bytes) -> char* {
    char* r = ws + off;
    off = (off + bytes + 255) & ~(size_t)255;
    return r;
  };
  // bf16 weight buffers
  __bf16* wfm  = (__bf16*)alloc(256 * 128 * 2);
  __bf16* wke3 = (__bf16*)alloc(256 * 128 * 2);
  __bf16* wq   = (__bf16*)alloc((size_t)6 * 65536 * 2);
  __bf16* wk   = (__bf16*)alloc((size_t)6 * 65536 * 2);
  __bf16* wv   = (__bf16*)alloc((size_t)6 * 65536 * 2);
  __bf16* wm   = (__bf16*)alloc((size_t)6 * 65536 * 2);
  __bf16* wml1 = (__bf16*)alloc((size_t)6 * 262144 * 2);
  __bf16* wml2 = (__bf16*)alloc((size_t)6 * 131072 * 2);
  __bf16* wfp  = (__bf16*)alloc(65536 * 2);
  __bf16* w1t  = (__bf16*)alloc((size_t)9 * 64 * 256 * 2);
  __bf16* w2t  = (__bf16*)alloc((size_t)9 * 128 * 64 * 2);
  // activations
  __bf16* gbuf   = (__bf16*)alloc((size_t)NB * 128 * NPT * 2);
  __bf16* x0buf  = (__bf16*)alloc((size_t)NB * 3 * NPT * 2);
  __bf16* e1     = (__bf16*)alloc((size_t)NB * 32 * NPT * 2);
  __bf16* e2     = (__bf16*)alloc((size_t)NB * 64 * NPT * 2);
  __bf16* e3     = (__bf16*)alloc((size_t)NB * 128 * NPT * 2);
  float*  gf     = (float*) alloc((size_t)NB * 256 * NPT * 4);
  float*  desc   = (float*) alloc((size_t)NB * 256 * NPT * 4);
  __bf16* ybuf   = (__bf16*)alloc((size_t)NB * 512 * NPT * 2);  // [desc | msg]
  __bf16* tbuf   = (__bf16*)alloc((size_t)NB * 512 * NPT * 2);
  __bf16* qbuf   = (__bf16*)alloc((size_t)NB * 256 * NPT * 2);
  __bf16* kbuf   = (__bf16*)alloc((size_t)NB * 256 * NPT * 2);
  __bf16* vbuf   = (__bf16*)alloc((size_t)NB * 256 * NPT * 2);
  __bf16* msgbuf = (__bf16*)alloc((size_t)NB * 256 * NPT * 2);
  __bf16* probs  = (__bf16*)alloc((size_t)NB * 4 * NPT * NPT * 2);
  __bf16* d0b    = (__bf16*)alloc((size_t)NB * 256 * NPT * 2);
  __bf16* zb     = (__bf16*)alloc((size_t)NB * 256 * HW2 * 2);
  __bf16* ycnn   = (__bf16*)alloc((size_t)NB * 64 * HW2 * 2);

  auto cvt = [&](const float* s, __bf16* d, long long n) {
    k_cvt<<<(unsigned)((n + 255) / 256), 256, 0, stream>>>(s, d, n);
  };
  cvt(fm_w, wfm, 256 * 128);
  cvt(ke_w3, wke3, 256 * 128);
  cvt(at_qw, wq, 6LL * 65536); cvt(at_kw, wk, 6LL * 65536);
  cvt(at_vw, wv, 6LL * 65536); cvt(at_mw, wm, 6LL * 65536);
  cvt(ml_w1, wml1, 6LL * 262144); cvt(ml_w2, wml2, 6LL * 131072);
  cvt(fp_w, wfp, 65536);
  k_rearr<<<(9 * 64 * 256 + 255) / 256, 256, 0, stream>>>(dc_w1, w1t, 64, 256);
  k_rearr<<<(9 * 128 * 64 + 255) / 256, 256, 0, stream>>>(dc_w2, w2t, 128, 64);

  auto mk = []() { GemmP p{}; p.alpha = 1.f; p.HZ = 1; p.ldc_n = 1; return p; };

  // gather + feature projection: gf = fm_w * gathered + fm_b
  k_gather<<<(NB * 128 * NPT + 255) / 256, 256, 0, stream>>>(feature, kp, gbuf);
  {
    GemmP p = mk();
    p.A = wfm; p.Bm = gbuf; p.Cf = gf; p.bias = fm_b;
    p.M = 256; p.N = NPT; p.K = 128; p.lda = 128; p.ldb = NPT; p.ldc_m = NPT;
    p.sBb = 128 * NPT; p.sCfb = 256 * NPT;
    gemm(stream, p, NB, 0);
  }
  // keypoint encoder
  k_keinput<<<(NB * NPT + 255) / 256, 256, 0, stream>>>(kp, x0buf);
  k_lin_small<<<(NB * 32 * NPT + 255) / 256, 256, 0, stream>>>(x0buf, ke_w0, ke_b0, ke_g0, ke_be0, e1, 3, 32);
  k_lin_small<<<(NB * 64 * NPT + 255) / 256, 256, 0, stream>>>(e1, ke_w1, ke_b1, ke_g1, ke_be1, e2, 32, 64);
  k_lin_small<<<(NB * 128 * NPT + 255) / 256, 256, 0, stream>>>(e2, ke_w2, ke_b2, ke_g2, ke_be2, e3, 64, 128);
  {  // desc = gf + ke_w3*e3 + ke_b3  (f32 + bf16 copy into ybuf desc-half)
    GemmP p = mk();
    p.A = wke3; p.Bm = e3; p.Cf = desc; p.Cb = ybuf; p.bias = ke_b3; p.resid = gf;
    p.M = 256; p.N = NPT; p.K = 128; p.lda = 128; p.ldb = NPT; p.ldc_m = NPT;
    p.sBb = 128 * NPT; p.sCfb = 256 * NPT; p.sCbb = 512 * NPT; p.sRb = 256 * NPT;
    gemm(stream, p, NB, 0);
  }

  // attentional GNN
  for (int i = 0; i < 6; ++i) {
    const __bf16* pw[3] = {wq + (size_t)i * 65536, wk + (size_t)i * 65536, wv + (size_t)i * 65536};
    const float*  pb[3] = {at_qb + i * 256, at_kb + i * 256, at_vb + i * 256};
    __bf16* pd[3] = {qbuf, kbuf, vbuf};
    for (int x = 0; x < 3; ++x) {
      GemmP p = mk();
      p.A = pw[x]; p.Bm = ybuf; p.Cb = pd[x]; p.bias = pb[x];
      p.M = 256; p.N = NPT; p.K = 256; p.lda = 256; p.ldb = NPT; p.ldc_m = NPT;
      p.sBb = 512 * NPT; p.sCbb = 256 * NPT;
      gemm(stream, p, NB, 0);
    }
    {  // scores[n,m] = (Q_h^T K_h)/8 per (b,h); channel c = d*4+h -> lda = 4N
      GemmP p = mk();
      p.A = qbuf; p.Bm = kbuf; p.Cb = probs;
      p.M = NPT; p.N = NPT; p.K = 64; p.lda = 4 * NPT; p.ldb = 4 * NPT; p.ldc_m = NPT;
      p.HZ = 4; p.sAb = 256 * NPT; p.sAh = NPT; p.sBb = 256 * NPT; p.sBh = NPT;
      p.sCbb = 4LL * NPT * NPT; p.sCh = (long long)NPT * NPT;
      p.alpha = 0.125f;
      gemm(stream, p, NB * 4, 1);  // A-transposed variant
    }
    k_softmax<<<NB * 4 * NPT / 8, 256, 0, stream>>>(probs);
    {  // msg^T[n,d] = P * V_h^T ; stored interleaved back into [B,256,N]
      GemmP p = mk();
      p.A = probs; p.Bm = vbuf; p.Cb = msgbuf;
      p.M = NPT; p.N = 64; p.K = NPT; p.lda = NPT; p.ldb = 4 * NPT;
      p.ldc_m = 1; p.ldc_n = 4 * NPT;
      p.HZ = 4; p.sAb = 4LL * NPT * NPT; p.sAh = (long long)NPT * NPT;
      p.sBb = 256 * NPT; p.sBh = NPT; p.sCbb = 256 * NPT; p.sCh = NPT;
      gemm(stream, p, NB * 4, 2);  // B-transposed variant
    }
    {  // merge -> second half of ybuf
      GemmP p = mk();
      p.A = wm + (size_t)i * 65536; p.Bm = msgbuf; p.Cb = ybuf + 256 * NPT;
      p.bias = at_mb + i * 256;
      p.M = 256; p.N = NPT; p.K = 256; p.lda = 256; p.ldb = NPT; p.ldc_m = NPT;
      p.sBb = 256 * NPT; p.sCbb = 512 * NPT;
      gemm(stream, p, NB, 0);
    }
    {  // MLP layer 1: relu(BN(W1*[desc|msg] + b1))
      GemmP p = mk();
      p.A = wml1 + (size_t)i * 262144; p.Bm = ybuf; p.Cb = tbuf;
      p.bias = ml_b1 + i * 512; p.gamma = ml_g1 + i * 512; p.beta = ml_be1 + i * 512;
      p.relu = 1;
      p.M = 512; p.N = NPT; p.K = 512; p.lda = 512; p.ldb = NPT; p.ldc_m = NPT;
      p.sBb = 512 * NPT; p.sCbb = 512 * NPT;
      gemm(stream, p, NB, 0);
    }
    {  // MLP layer 2 + residual -> desc (f32) and ybuf desc-half (bf16)
      GemmP p = mk();
      p.A = wml2 + (size_t)i * 131072; p.Bm = tbuf; p.Cf = desc; p.Cb = ybuf;
      p.bias = ml_b2 + i * 256; p.resid = desc;
      p.M = 256; p.N = NPT; p.K = 512; p.lda = 512; p.ldb = NPT; p.ldc_m = NPT;
      p.sBb = 512 * NPT; p.sCfb = 256 * NPT; p.sCbb = 512 * NPT; p.sRb = 256 * NPT;
      gemm(stream, p, NB, 0);
    }
  }

  // final projection: desc0 -> d_out part 2 (f32) + bf16 copy for scatter
  float* out_desc0 = (float*)d_out + (size_t)NB * 128 * HW2;
  {
    GemmP p = mk();
    p.A = wfp; p.Bm = ybuf; p.Cf = out_desc0; p.Cb = d0b; p.bias = fp_b;
    p.M = 256; p.N = NPT; p.K = 256; p.lda = 256; p.ldb = NPT; p.ldc_m = NPT;
    p.sBb = 512 * NPT; p.sCfb = 256 * NPT; p.sCbb = 256 * NPT;
    gemm(stream, p, NB, 0);
  }

  // scatter onto grid, then diffusion CNN
  long long zwords = (long long)NB * 256 * HW2 / 2;  // bf16 pairs as u32
  k_zero32<<<(unsigned)((zwords + 255) / 256), 256, 0, stream>>>((unsigned*)zb, zwords);
  k_scatter<<<(NB * NPT + 255) / 256, 256, 0, stream>>>(kp, d0b, zb);

  {
    ConvP c{};
    c.in = zb; c.wt = w1t; c.bias = dc_b1; c.gamma = dc_g; c.beta = dc_be;
    c.outb = ycnn; c.CIN = 256; c.CO = 64; c.relu = 1;
    k_conv3<<<dim3(4, 256, NB), 128, 0, stream>>>(c);
  }
  {
    ConvP c{};
    c.in = ycnn; c.wt = w2t; c.bias = dc_b2;
    c.outf = (float*)d_out; c.CIN = 64; c.CO = 128; c.relu = 0;
    k_conv3<<<dim3(4, 256, NB * 2), 128, 0, stream>>>(c);
  }
  (void)in_sizes; (void)n_in; (void)out_size; (void)ws_size;
}
